// MLMA_77790447665619
// MI455X (gfx1250) — compile-verified, matrix operand order checked by reading, not run on hardware
//
#include <hip/hip_runtime.h>

#define BB  32
#define LL  1024
#define CC  1024
#define DD1 128
#define DD2 32

typedef __attribute__((ext_vector_type(16))) __bf16 v16bf;
typedef __attribute__((ext_vector_type(8)))  float  v8f;
typedef __attribute__((ext_vector_type(4)))  unsigned int u32x4;
typedef __attribute__((ext_vector_type(4)))  int  i32x4;
typedef __attribute__((ext_vector_type(8)))  int  i32x8;

union FragU { v16bf v; uint4 q[2]; };

// Load one 16x32 bf16 WMMA fragment (A from row-major, or B from "column-major",
// i.e. memory where each of the 16 rows/cols has its 32 K-values contiguous with
// stride `ld` elements). ISA layout: lanes 0-15 hold K 0-7 (elems 0-7) and
// K 16-23 (elems 8-15); lanes 16-31 hold K 8-15 and K 24-31.
__device__ __forceinline__ v16bf load_frag16(const __bf16* base, int ld) {
  const int lane = threadIdx.x & 31;
  const int r    = lane & 15;
  const int kb   = (lane >> 4) << 3;      // 0 or 8
  const __bf16* p = base + (size_t)r * ld + kb;
  FragU f;
  f.q[0] = *(const uint4*)(p);            // K kb .. kb+7
  f.q[1] = *(const uint4*)(p + 16);       // K kb+16 .. kb+23
  return f.v;
}

__device__ __forceinline__ v8f wmma_bf16(v16bf a, v16bf b, v8f c) {
  return __builtin_amdgcn_wmma_f32_16x16x32_bf16(false, a, false, b, (short)0, c,
                                                 false, false);
}

// ------------------------------------------------------------------ TDM helpers
#if defined(__AMDGCN__) && __has_builtin(__builtin_amdgcn_tensor_load_to_lds) && \
    __has_builtin(__builtin_amdgcn_s_wait_tensorcnt)
#define HAVE_TDM 1
#else
#define HAVE_TDM 0
#endif

#if HAVE_TDM
// D# group1 for a 2D bf16 tile: data_size=2B, tile 32(dim0) x tile_d1(dim1),
// tensor_dim0/1 and dim0 stride in elements. (ISA ch.8, group1 bit packing.)
__device__ __forceinline__ i32x8 tdm_desc_g1_2d(int tdim0, int tdim1,
                                                int tile_d0, int tile_d1,
                                                int stride0) {
  i32x8 g;
  g[0] = 1 << 16;                                        // data_size = 2 bytes
  g[1] = (tdim0 & 0xFFFF) << 16;                         // tensor_dim0[15:0]
  g[2] = ((tdim0 >> 16) & 0xFFFF) | ((tdim1 & 0xFFFF) << 16);
  g[3] = ((tdim1 >> 16) & 0xFFFF) | ((tile_d0 & 0xFFFF) << 16);
  g[4] = tile_d1 & 0xFFFF;                               // tile_dim1; tile_dim2=0
  g[5] = stride0;                                        // dim0 stride [31:0]
  g[6] = 0;                                              // stride hi / dim1 stride
  g[7] = 0;
  return g;
}

// Issue one TDM 2D tile load: global tile start -> LDS byte offset.
__device__ __forceinline__ void tdm_load_2d(unsigned lds_byte_off,
                                            const void* gptr, i32x8 g1) {
  unsigned long long ga = (unsigned long long)(size_t)gptr;
  u32x4 g0;
  g0.x = 1u;                                   // count=1 (valid descriptor)
  g0.y = lds_byte_off;                         // lds_addr
  g0.z = (unsigned)(ga & 0xFFFFFFFFu);         // global_addr[31:0]
  g0.w = (unsigned)(ga >> 32) | (2u << 30);    // global_addr[56:32] | type=2
  i32x4 z4 = {0, 0, 0, 0};
#if __clang_major__ >= 23
  i32x8 z8 = {0, 0, 0, 0, 0, 0, 0, 0};
  __builtin_amdgcn_tensor_load_to_lds(g0, g1, z4, z4, z8, 0);
#else
  __builtin_amdgcn_tensor_load_to_lds(g0, g1, z4, z4, 0);
#endif
}
#define LDS_OFF(p) ((unsigned)(size_t)(const void*)(p))
#endif

// ---------------------------------------------------------------- convert
__global__ void cvt_bf16_kernel(const float* __restrict__ in,
                                __bf16* __restrict__ out, int n) {
  for (int i = blockIdx.x * blockDim.x + threadIdx.x; i < n;
       i += gridDim.x * blockDim.x)
    out[i] = (__bf16)in[i];
}

// ------------------------------------------------- fused conv1+relu+conv2+relu
// grid = B * (L/16), block = 64 (2 waves). Produces y[b][l][32] bf16 (row major),
// which serves as both A (rows=l) and column-major B (rows=m) for the gram WMMA.
__global__ __launch_bounds__(64) void conv_fused_kernel(
    const __bf16* __restrict__ Abf,  // inpt bf16 [B][L][C]
    const __bf16* __restrict__ W1bf, // [D1][C]
    const float*  __restrict__ b1,
    const __bf16* __restrict__ W2bf, // [D2][D1]
    const float*  __restrict__ b2,
    __bf16* __restrict__ y)          // [B][L][D2]
{
  const int wg = blockIdx.x;
  const int b  = wg >> 6;
  const int l0 = (wg & 63) * 16;
  const int wv   = threadIdx.x >> 5;
  const int lane = threadIdx.x & 31;
  const int n  = lane & 15;
  const int hi = lane >> 4;

  __shared__ __bf16 x1[16][DD1];     // 4 KB intermediate tile

  const __bf16* Ab = Abf + ((size_t)b * LL + l0) * CC;
  const __bf16* Wb = W1bf + (size_t)(wv * 4 * 16) * CC;

  // conv1: each wave handles 4 d-tiles (64 output channels), K = 1024.
  // Distinct fragment values per step -> scheduler can overlap loads & wmma.
  v8f acc[4];
  for (int i = 0; i < 4; ++i) acc[i] = {};
  for (int kc = 0; kc < CC / 32; ++kc) {
    v16bf a  = load_frag16(Ab + kc * 32, CC);
    v16bf w0 = load_frag16(Wb + (size_t)(0 * 16) * CC + kc * 32, CC);
    v16bf w1 = load_frag16(Wb + (size_t)(1 * 16) * CC + kc * 32, CC);
    v16bf w2 = load_frag16(Wb + (size_t)(2 * 16) * CC + kc * 32, CC);
    v16bf w3 = load_frag16(Wb + (size_t)(3 * 16) * CC + kc * 32, CC);
    acc[0] = wmma_bf16(a, w0, acc[0]);
    acc[1] = wmma_bf16(a, w1, acc[1]);
    acc[2] = wmma_bf16(a, w2, acc[2]);
    acc[3] = wmma_bf16(a, w3, acc[3]);
  }
  for (int i = 0; i < 4; ++i) {
    const int d   = (wv * 4 + i) * 16 + n;
    const float bias = b1[d];
    for (int r = 0; r < 8; ++r) {
      float v = acc[i][r] + bias;
      x1[hi * 8 + r][d] = (__bf16)(v > 0.f ? v : 0.f);
    }
  }
  __syncthreads();

  // conv2: wave wv handles e-tile wv (16 channels), K = 128
  v8f acc2 = {};
  for (int kc = 0; kc < DD1 / 32; ++kc) {
    v16bf a = load_frag16(&x1[0][0] + kc * 32, DD1);
    v16bf w = load_frag16(W2bf + (size_t)(wv * 16) * DD1 + kc * 32, DD1);
    acc2 = wmma_bf16(a, w, acc2);
  }
  const int e = wv * 16 + n;
  const float bias2 = b2[e];
  __bf16* yb = y + ((size_t)b * LL + l0) * DD2;
  for (int r = 0; r < 8; ++r) {
    float v = acc2[r] + bias2;
    yb[(size_t)(hi * 8 + r) * DD2 + e] = (__bf16)(v > 0.f ? v : 0.f);
  }
}

// -------------------------------------------- gram = softmax(y yT), transposed
// grid = B * (L/16), block = 256 (8 waves). Each block: 16 gram rows x 1024 cols.
// K = 32 -> exactly one v_wmma per 16x16 tile. Softmax is done in the WMMA
// C-register layout with __shfl_xor half-wave reductions + LDS cross-wave.
// Output: gramT[b][m][k] bf16 (so kernel 3's B loads are contiguous in K).
__global__ __launch_bounds__(256) void gram_softmax_kernel(
    const __bf16* __restrict__ y, __bf16* __restrict__ gramT)
{
  const int wg = blockIdx.x;
  const int b  = wg >> 6;
  const int l0 = (wg & 63) * 16;
  const int wv   = threadIdx.x >> 5;
  const int lane = threadIdx.x & 31;
  const int n  = lane & 15;
  const int hi = lane >> 4;

  __shared__ float sRed[8][16];
  __shared__ float sStat[16];

  const __bf16* yb = y + (size_t)b * LL * DD2;
  v16bf a = load_frag16(yb + (size_t)l0 * DD2, DD2);

  v8f acc[8];
  for (int t = 0; t < 8; ++t) {
    const int m0 = wv * 128 + t * 16;
    v16bf bm = load_frag16(yb + (size_t)m0 * DD2, DD2);
    v8f c = {};
    acc[t] = wmma_bf16(a, bm, c);
  }

  // --- row max (row = hi*8 + r) ---
  float pm[8];
  for (int r = 0; r < 8; ++r) {
    float m = -1e30f;
    for (int t = 0; t < 8; ++t) m = fmaxf(m, acc[t][r]);
    pm[r] = m;
  }
  for (int mk = 1; mk <= 8; mk <<= 1)
    for (int r = 0; r < 8; ++r)
      pm[r] = fmaxf(pm[r], __shfl_xor(pm[r], mk, 32));
  if (n == 0)
    for (int r = 0; r < 8; ++r) sRed[wv][hi * 8 + r] = pm[r];
  __syncthreads();
  if (threadIdx.x < 16) {
    float m = -1e30f;
    for (int w = 0; w < 8; ++w) m = fmaxf(m, sRed[w][threadIdx.x]);
    sStat[threadIdx.x] = m;
  }
  __syncthreads();
  float rowmax[8];
  for (int r = 0; r < 8; ++r) rowmax[r] = sStat[hi * 8 + r];

  // --- exp + row sum ---
  float ps[8];
  for (int r = 0; r < 8; ++r) ps[r] = 0.f;
  for (int t = 0; t < 8; ++t)
    for (int r = 0; r < 8; ++r) {
      float e = __expf(acc[t][r] - rowmax[r]);
      acc[t][r] = e;
      ps[r] += e;
    }
  for (int mk = 1; mk <= 8; mk <<= 1)
    for (int r = 0; r < 8; ++r) ps[r] += __shfl_xor(ps[r], mk, 32);
  __syncthreads();
  if (n == 0)
    for (int r = 0; r < 8; ++r) sRed[wv][hi * 8 + r] = ps[r];
  __syncthreads();
  if (threadIdx.x < 16) {
    float s = 0.f;
    for (int w = 0; w < 8; ++w) s += sRed[w][threadIdx.x];
    sStat[threadIdx.x] = s;
  }
  __syncthreads();
  float rinv[8];
  for (int r = 0; r < 8; ++r) rinv[r] = 1.f / sStat[hi * 8 + r];

  // --- normalize + transposed bf16 store: gramT[b][m][l0 + hi*8 + r] ---
  __bf16* g = gramT + (size_t)b * LL * LL;
  for (int t = 0; t < 8; ++t) {
    const int m = wv * 128 + t * 16 + n;
    union { __bf16 h[8]; uint4 q; } u;
    for (int r = 0; r < 8; ++r) u.h[r] = (__bf16)(acc[t][r] * rinv[r]);
    *(uint4*)(g + (size_t)m * LL + l0 + hi * 8) = u.q;
  }
}

// ----------------------------------- embed = (inpt @ gram)^T, the 68.7 GFLOP GEMM
// grid = B * 8 * 8 (128x128 output blocks), block = 256 (8 waves).
// K-panels (A 128x32 bf16, B 128x32 bf16) are DMA'd into double-buffered LDS by
// the Tensor Data Mover (wave 0 issues the descriptors; TENSORcnt tracks them),
// so all VALU/VMEM issue slots go to v_wmma + ds_load. Falls back to manual
// b128 staging when the TDM builtin is unavailable (e.g. host pass).
__global__ __launch_bounds__(256) void embed_gemm_kernel(
    const __bf16* __restrict__ Abf,   // inpt bf16 [B][L][C]
    const __bf16* __restrict__ gT,    // gramT bf16 [B][m][k]
    float* __restrict__ out)          // [B][m][l] f32
{
  const int wg  = blockIdx.x;
  const int b   = wg >> 6;
  const int rem = wg & 63;
  const int l0  = (rem >> 3) * 128;
  const int m0  = (rem & 7) * 128;
  const int wv   = threadIdx.x >> 5;
  const int lane = threadIdx.x & 31;
  const int n  = lane & 15;
  const int hi = lane >> 4;

  __shared__ __bf16 As[2][128 * 32];   // 2 x 8 KB
  __shared__ __bf16 Bs[2][128 * 32];   // 2 x 8 KB

  const __bf16* Ab = Abf + ((size_t)b * LL + l0) * CC;
  const __bf16* Bb = gT  + ((size_t)b * LL + m0) * LL;

  v8f acc[8];
  for (int j = 0; j < 8; ++j) acc[j] = {};

#if HAVE_TDM
  const bool issuer = (threadIdx.x < 32);   // wave 0 issues TDM descriptors
  // Same 2D descriptor shape for both matrices: 1024x1024 bf16 tensor,
  // 32(K) x 128(row) tile, row stride 1024 elements.
  i32x8 g1 = tdm_desc_g1_2d(CC, LL, 32, 128, CC);
  if (issuer) {
    tdm_load_2d(LDS_OFF(&As[0][0]), Ab, g1);
    tdm_load_2d(LDS_OFF(&Bs[0][0]), Bb, g1);
  }
  for (int kc = 0; kc < CC / 32; ++kc) {
    if (issuer) {
      if (kc + 1 < CC / 32) {
        const int nb = (kc + 1) & 1;
        tdm_load_2d(LDS_OFF(&As[nb][0]), Ab + (kc + 1) * 32, g1);
        tdm_load_2d(LDS_OFF(&Bs[nb][0]), Bb + (kc + 1) * 32, g1);
        __builtin_amdgcn_s_wait_tensorcnt(2);  // panel kc complete
      } else {
        __builtin_amdgcn_s_wait_tensorcnt(0);
      }
    }
    __syncthreads();
    const __bf16* Asb = As[kc & 1];
    const __bf16* Bsb = Bs[kc & 1];
    v16bf a = load_frag16(Asb + (size_t)(wv * 16) * 32, 32);
    for (int j = 0; j < 8; ++j) {
      v16bf bm = load_frag16(Bsb + (size_t)(j * 16) * 32, 32);
      acc[j] = wmma_bf16(a, bm, acc[j]);
    }
    __syncthreads();
  }
#else
  const int t = threadIdx.x;
  for (int kc = 0; kc < CC / 32; ++kc) {
    for (int q = t; q < 512; q += 256) {
      const int row = q >> 2, c8 = (q & 3) * 8;
      ((uint4*)&As[0][0])[q] = *(const uint4*)(Ab + (size_t)row * CC + kc * 32 + c8);
      ((uint4*)&Bs[0][0])[q] = *(const uint4*)(Bb + (size_t)row * LL + kc * 32 + c8);
    }
    if (kc + 1 < CC / 32) {   // prefetch next panel -> global_prefetch_b8
      __builtin_prefetch(Ab + (size_t)(t >> 1) * CC + (kc + 1) * 32, 0, 1);
      __builtin_prefetch(Bb + (size_t)(t >> 1) * LL + (kc + 1) * 32, 0, 1);
    }
    __syncthreads();
    v16bf a = load_frag16(&As[0][0] + (size_t)(wv * 16) * 32, 32);
    for (int j = 0; j < 8; ++j) {
      v16bf bm = load_frag16(&Bs[0][0] + (size_t)(j * 16) * 32, 32);
      acc[j] = wmma_bf16(a, bm, acc[j]);
    }
    __syncthreads();
  }
#endif

  // store: out[b][m][l]; a lane's 8 acc elements are 8 consecutive l -> 2x b128
  float* ob = out + (size_t)b * LL * LL;
  const int l = l0 + wv * 16 + hi * 8;
  for (int j = 0; j < 8; ++j) {
    const int m = m0 + j * 16 + n;
    float* dst = ob + (size_t)m * LL + l;
    *(float4*)(dst)     = make_float4(acc[j][0], acc[j][1], acc[j][2], acc[j][3]);
    *(float4*)(dst + 4) = make_float4(acc[j][4], acc[j][5], acc[j][6], acc[j][7]);
  }
}

extern "C" void kernel_launch(void* const* d_in, const int* in_sizes, int n_in,
                              void* d_out, int out_size, void* d_ws, size_t ws_size,
                              hipStream_t stream) {
  const float* inpt = (const float*)d_in[0];
  const float* W1   = (const float*)d_in[1];
  const float* b1   = (const float*)d_in[2];
  const float* W2   = (const float*)d_in[3];
  const float* b2   = (const float*)d_in[4];
  float* out        = (float*)d_out;

  unsigned char* ws = (unsigned char*)d_ws;
  const size_t nInpt = (size_t)BB * LL * CC;
  const size_t nW1   = (size_t)DD1 * CC;
  const size_t nW2   = (size_t)DD2 * DD1;
  __bf16* inpt_bf = (__bf16*)(ws);
  __bf16* W1_bf   = (__bf16*)(ws + nInpt * 2);
  __bf16* W2_bf   = (__bf16*)(ws + nInpt * 2 + nW1 * 2);
  __bf16* y_bf    = (__bf16*)(ws + nInpt * 2 + nW1 * 2 + nW2 * 2);
  __bf16* gramT   = (__bf16*)(ws + nInpt * 2 + nW1 * 2 + nW2 * 2 +
                              (size_t)BB * LL * DD2 * 2);

  cvt_bf16_kernel<<<2048, 256, 0, stream>>>(inpt, inpt_bf, (int)nInpt);
  cvt_bf16_kernel<<<64,   256, 0, stream>>>(W1,   W1_bf,   (int)nW1);
  cvt_bf16_kernel<<<8,    256, 0, stream>>>(W2,   W2_bf,   (int)nW2);

  conv_fused_kernel<<<BB * (LL / 16), 64, 0, stream>>>(inpt_bf, W1_bf, b1,
                                                       W2_bf, b2, y_bf);
  gram_softmax_kernel<<<BB * (LL / 16), 256, 0, stream>>>(y_bf, gramT);
  embed_gemm_kernel<<<BB * 8 * 8, 256, 0, stream>>>(inpt_bf, gramT, out);
}
